// RotarySelfAttention_41051297415426
// MI455X (gfx1250) — matrix operand order repaired and checked by reading, not compiled, not run on hardware
//
#include <hip/hip_runtime.h>
#include <stdint.h>

#define BB  16
#define TNN 16
#define DD  1024
#define HH  16
#define DHH 64
#define LPP 2048

typedef __attribute__((ext_vector_type(16))) __bf16 v16bf;
typedef __attribute__((ext_vector_type(8)))  float  v8f;
typedef __attribute__((ext_vector_type(4)))  float  f32x4;
typedef __attribute__((ext_vector_type(4)))  unsigned int u32x4;
typedef __attribute__((__vector_size__(16))) int vsi4;   // matches builtin param
typedef unsigned short ushort_t;

union FragAB {
  v16bf bf;
  u32x4 q[2];
};

__device__ __forceinline__ ushort_t f2bf(float f) {
  unsigned x = __builtin_bit_cast(unsigned, f);
  x += 0x7FFFu + ((x >> 16) & 1u);          // round-to-nearest-even
  return (ushort_t)(x >> 16);
}
__device__ __forceinline__ unsigned pk2(float a, float b) {
  return (unsigned)f2bf(a) | ((unsigned)f2bf(b) << 16);
}
__device__ __forceinline__ v8f wmma_bf16(const FragAB& a, const FragAB& b, v8f c) {
  return __builtin_amdgcn_wmma_f32_16x16x32_bf16(false, a.bf, false, b.bf,
                                                 (short)0, c, false, false);
}

// ---- CDNA5 async global->LDS (ASYNCcnt-tracked DMA, no VGPR landing zone) ----
__device__ __forceinline__ void async_b128(const void* g, void* l) {
#if __has_builtin(__builtin_amdgcn_global_load_async_to_lds_b128)
  __builtin_amdgcn_global_load_async_to_lds_b128(
      (__attribute__((address_space(1))) vsi4*)(vsi4*)(void*)g,
      (__attribute__((address_space(3))) vsi4*)(vsi4*)l,
      0, 0);
#else
  unsigned loff = (unsigned)(uintptr_t)(__attribute__((address_space(3))) void*)l;
  asm volatile("global_load_async_to_lds_b128 %0, %1, off"
               :: "v"(loff), "v"(g) : "memory");
#endif
}
__device__ __forceinline__ void wait_async_le32() {
#if __has_builtin(__builtin_amdgcn_s_wait_asynccnt)
  __builtin_amdgcn_s_wait_asynccnt(32);
#else
  asm volatile("s_wait_asynccnt 0x20" ::: "memory");
#endif
}
__device__ __forceinline__ void wait_ds0() {
  asm volatile("s_wait_dscnt 0x0" ::: "memory");
}

// ---------------------------------------------------------------- cvt fp32->bf16
__global__ void __launch_bounds__(256) cvt_kernel(const float* __restrict__ src,
                                                  ushort_t* __restrict__ dst, int n4) {
  int i = blockIdx.x * blockDim.x + threadIdx.x;
  if (i >= n4) return;
  f32x4 v = *(const f32x4*)(src + (size_t)i * 4);
  unsigned* d = (unsigned*)(dst + (size_t)i * 4);
  d[0] = pk2(v.x, v.y);
  d[1] = pk2(v.z, v.w);
}

// ---------------------------------------------------------------- QKV projection
__global__ void __launch_bounds__(32) qkv_gemm_kernel(
    const ushort_t* __restrict__ xbf,
    const ushort_t* __restrict__ wq, const ushort_t* __restrict__ wk,
    const ushort_t* __restrict__ wv,
    const float* __restrict__ bq, const float* __restrict__ bk,
    const float* __restrict__ bv,
    float* __restrict__ qraw, float* __restrict__ kraw, float* __restrict__ vraw)
{
  const int l = threadIdx.x, lm = l & 15, lg = l >> 4;
  const int n0 = blockIdx.x * 16, m0 = blockIdx.y * 16, mat = blockIdx.z;
  const ushort_t* W  = (mat == 0) ? wq : (mat == 1) ? wk : wv;
  const float*  bias = (mat == 0) ? bq : (mat == 1) ? bk : bv;
  float*        out  = (mat == 0) ? qraw : (mat == 1) ? kraw : vraw;

  v8f acc = {};
  const ushort_t* arow = xbf + (size_t)(m0 + lm) * DD;
  const ushort_t* brow = W   + (size_t)(n0 + lm) * DD;
  for (int k0 = 0; k0 < DD; k0 += 32) {
    FragAB af, bf_;
    af.q[0]  = *(const u32x4*)(arow + k0 + lg * 8);
    af.q[1]  = *(const u32x4*)(arow + k0 + 16 + lg * 8);
    bf_.q[0] = *(const u32x4*)(brow + k0 + lg * 8);
    bf_.q[1] = *(const u32x4*)(brow + k0 + 16 + lg * 8);
    acc = wmma_bf16(af, bf_, acc);
  }
  const float bv_ = bias[n0 + lm];
  const int col = n0 + lm, h = col >> 6, dh = col & 63;
  #pragma unroll
  for (int r = 0; r < 8; ++r) {
    const int row = m0 + r + lg * 8;
    const int b = row >> 4, t = row & 15;
    out[(size_t)(((b * HH + h) * TNN + t) * DHH) + dh] = acc[r] + bv_;
  }
}

// ---------------------------------------------------------------- rotary (q,k in-place)
__global__ void __launch_bounds__(256) rotary_kernel(
    float* __restrict__ q, float* __restrict__ k,
    const float* __restrict__ cosb, const float* __restrict__ sinb)
{
  int i = blockIdx.x * blockDim.x + threadIdx.x;       // B*H*TN*32
  if (i >= BB * HH * TNN * 32) return;
  const int dh = i & 31, t = (i >> 5) & 15, bh = i >> 9;
  const size_t base = (size_t)(bh * TNN + t) * DHH;
  const float c1 = cosb[t * DHH + dh],      s1 = sinb[t * DHH + dh];
  const float c2 = cosb[t * DHH + dh + 32], s2 = sinb[t * DHH + dh + 32];
  const float q1 = q[base + dh], q2 = q[base + dh + 32];
  q[base + dh]      = q1 * c1 - q2 * s1;
  q[base + dh + 32] = q2 * c2 + q1 * s2;
  const float k1 = k[base + dh], k2 = k[base + dh + 32];
  k[base + dh]      = k1 * c1 - k2 * s1;
  k[base + dh + 32] = k2 * c2 + k1 * s2;
}

// ---------------------------------------------------------------- flash attention
// 1 wave / (b,h); async double-buffered KV staging (fp32 LDS) -> bf16 fragments.
__global__ void __launch_bounds__(32) attn_kernel(
    const float* __restrict__ qrot, const float* __restrict__ krot,
    const float* __restrict__ vraw, const float* __restrict__ pastk,
    const float* __restrict__ pastv, const int* __restrict__ plen_arr,
    const int* __restrict__ ncnt_arr, ushort_t* __restrict__ attn_bf)
{
  __shared__ float    Kf[2][32 * DHH];  // async-landing K tiles (fp32)
  __shared__ float    Vf[2][32 * DHH];  // async-landing V tiles (fp32)
  __shared__ ushort_t Qs[TNN * DHH];    // 16 x 64 bf16
  __shared__ ushort_t Ks[32 * DHH];     // 32 keys x 64 bf16
  __shared__ ushort_t Vt[DHH * 32];     // transposed: [dh][key]
  __shared__ ushort_t Ps[TNN * 32];     // 16 x 32 probs

  const int l = threadIdx.x, lm = l & 15, lg = l >> 4;
  const int bh = blockIdx.x, b = bh >> 4, h = bh & 15;
  const int plen  = plen_arr[b];
  const int total = plen + ncnt_arr[b];
  const int ntiles = (total + 31) >> 5;

  // Issue one 32-key tile (this lane owns key = keybase + l). Addresses are
  // clamped so they are always dereferenceable; invalid keys are masked later.
  auto issue_tile = [&](int tile) {
    int kb = tile << 5;
    if (kb >= total) kb = (ntiles - 1) << 5;   // clamp redundant prefetch
    const int key = kb + l;
    const float *ksrc, *vsrc;
    if (key < plen) {
      ksrc = pastk + ((size_t)bh * LPP + key) * DHH;
      vsrc = pastv + ((size_t)bh * LPP + key) * DHH;
    } else {
      int t = key - plen; if (t > TNN - 1) t = TNN - 1; if (t < 0) t = 0;
      ksrc = krot + (size_t)(bh * TNN + t) * DHH;
      vsrc = vraw + (size_t)(bh * TNN + t) * DHH;
    }
    float* kdst = &Kf[tile & 1][l * DHH];
    float* vdst = &Vf[tile & 1][l * DHH];
    #pragma unroll
    for (int j = 0; j < DHH; j += 4) {          // 16+16 async b128 ops / tile
      async_b128(ksrc + j, kdst + j);
      async_b128(vsrc + j, vdst + j);
    }
  };

  { // stage Q as bf16 (lane: row lm, half lg)
    const float* src = qrot + (size_t)(bh * TNN + lm) * DHH + lg * 32;
    unsigned* dst = (unsigned*)(Qs + lm * DHH + lg * 32);
    #pragma unroll
    for (int j = 0; j < 32; j += 4) {
      f32x4 v = *(const f32x4*)(src + j);
      dst[j / 2]     = pk2(v.x, v.y);
      dst[j / 2 + 1] = pk2(v.z, v.w);
    }
  }

  issue_tile(0);                                // prologue prefetch

  v8f O0 = {}, O1 = {}, O2 = {}, O3 = {};
  float mrow[8], lrow[8];
  #pragma unroll
  for (int r = 0; r < 8; ++r) { mrow[r] = -1e38f; lrow[r] = 0.f; }

  for (int kt = 0; kt < ntiles; ++kt) {
    const int keybase = kt << 5;
    wait_ds0();                 // prior LDS reads of the target buffer are done
    issue_tile(kt + 1);         // prefetch next tile into the other buffer
    wait_async_le32();          // in-order completion => tile kt is resident

    { // convert fp32 tile -> bf16 Ks / transposed Vt (lane owns key row l)
      const float* kf = &Kf[kt & 1][l * DHH];
      const float* vf = &Vf[kt & 1][l * DHH];
      unsigned* kd = (unsigned*)(Ks + l * DHH);
      #pragma unroll 4
      for (int j = 0; j < DHH; j += 4) {
        f32x4 kv = *(const f32x4*)(kf + j);
        f32x4 vv = *(const f32x4*)(vf + j);
        kd[j / 2]     = pk2(kv.x, kv.y);
        kd[j / 2 + 1] = pk2(kv.z, kv.w);
        Vt[(j + 0) * 32 + l] = f2bf(vv.x);
        Vt[(j + 1) * 32 + l] = f2bf(vv.y);
        Vt[(j + 2) * 32 + l] = f2bf(vv.z);
        Vt[(j + 3) * 32 + l] = f2bf(vv.w);
      }
    }
    __syncthreads();

    // S = Q @ K^T  (two 16x16 key tiles, K-dim = DH in two 32-chunks)
    v8f S0 = {}, S1 = {};
    #pragma unroll
    for (int kk = 0; kk < DHH; kk += 32) {
      FragAB qa, k0f, k1f;
      const ushort_t* qr = Qs + lm * DHH + kk;
      qa.q[0] = *(const u32x4*)(qr + lg * 8);
      qa.q[1] = *(const u32x4*)(qr + 16 + lg * 8);
      const ushort_t* kr0 = Ks + lm * DHH + kk;
      k0f.q[0] = *(const u32x4*)(kr0 + lg * 8);
      k0f.q[1] = *(const u32x4*)(kr0 + 16 + lg * 8);
      const ushort_t* kr1 = Ks + (16 + lm) * DHH + kk;
      k1f.q[0] = *(const u32x4*)(kr1 + lg * 8);
      k1f.q[1] = *(const u32x4*)(kr1 + 16 + lg * 8);
      S0 = wmma_bf16(qa, k0f, S0);
      S1 = wmma_bf16(qa, k1f, S1);
    }

    // online softmax; C layout: lane covers key col lm (+16), rows r+lg*8
    const bool v0 = (keybase + lm) < total;
    const bool v1 = (keybase + 16 + lm) < total;
    float mloc[8];
    #pragma unroll
    for (int r = 0; r < 8; ++r) {
      float s0 = v0 ? S0[r] * 0.125f : -1e30f;    // 1/sqrt(64)
      float s1 = v1 ? S1[r] * 0.125f : -1e30f;
      S0[r] = s0; S1[r] = s1;
      mloc[r] = fmaxf(s0, s1);
    }
    #pragma unroll
    for (int off = 1; off < 16; off <<= 1)
      #pragma unroll
      for (int r = 0; r < 8; ++r)
        mloc[r] = fmaxf(mloc[r], __shfl_xor(mloc[r], off, 32));
    float alpha[8], rsum[8];
    #pragma unroll
    for (int r = 0; r < 8; ++r) {
      const float mnew = fmaxf(mrow[r], mloc[r]);
      alpha[r] = __expf(mrow[r] - mnew);
      mrow[r] = mnew;
      const float p0 = __expf(S0[r] - mnew);
      const float p1 = __expf(S1[r] - mnew);
      S0[r] = p0; S1[r] = p1;
      rsum[r] = p0 + p1;
    }
    #pragma unroll
    for (int off = 1; off < 16; off <<= 1)
      #pragma unroll
      for (int r = 0; r < 8; ++r)
        rsum[r] += __shfl_xor(rsum[r], off, 32);
    #pragma unroll
    for (int r = 0; r < 8; ++r) {
      lrow[r] = lrow[r] * alpha[r] + rsum[r];
      O0[r] *= alpha[r]; O1[r] *= alpha[r];
      O2[r] *= alpha[r]; O3[r] *= alpha[r];
    }
    #pragma unroll
    for (int r = 0; r < 8; ++r) {
      const int row = r + lg * 8;
      Ps[row * 32 + lm]      = f2bf(S0[r]);
      Ps[row * 32 + 16 + lm] = f2bf(S1[r]);
    }
    __syncthreads();

    // O += P @ V   (A = P 16x32, B = V 32xDH via transposed Vt)
    FragAB pa;
    const ushort_t* pr = Ps + lm * 32;
    pa.q[0] = *(const u32x4*)(pr + lg * 8);
    pa.q[1] = *(const u32x4*)(pr + 16 + lg * 8);
    #pragma unroll
    for (int j = 0; j < 4; ++j) {
      FragAB vb;
      const ushort_t* vr = Vt + (j * 16 + lm) * 32 + lg * 8;
      vb.q[0] = *(const u32x4*)(vr);
      vb.q[1] = *(const u32x4*)(vr + 16);
      if (j == 0) O0 = wmma_bf16(pa, vb, O0);
      else if (j == 1) O1 = wmma_bf16(pa, vb, O1);
      else if (j == 2) O2 = wmma_bf16(pa, vb, O2);
      else O3 = wmma_bf16(pa, vb, O3);
    }
  }

  #pragma unroll
  for (int r = 0; r < 8; ++r) lrow[r] = 1.f / lrow[r];
  ushort_t* dst = attn_bf + (size_t)(b * TNN) * DD + h * DHH;
  #pragma unroll
  for (int r = 0; r < 8; ++r) {
    const int t = r + lg * 8;
    ushort_t* drow = dst + (size_t)t * DD;
    drow[0  + lm] = f2bf(O0[r] * lrow[r]);
    drow[16 + lm] = f2bf(O1[r] * lrow[r]);
    drow[32 + lm] = f2bf(O2[r] * lrow[r]);
    drow[48 + lm] = f2bf(O3[r] * lrow[r]);
  }
}

// ---------------------------------------------------------------- output projection + mask
__global__ void __launch_bounds__(32) out_gemm_kernel(
    const ushort_t* __restrict__ abf, const ushort_t* __restrict__ wo,
    const float* __restrict__ bo, const int* __restrict__ ncnt,
    float* __restrict__ out)
{
  const int l = threadIdx.x, lm = l & 15, lg = l >> 4;
  const int n0 = blockIdx.x * 16, m0 = blockIdx.y * 16;
  v8f acc = {};
  const ushort_t* arow = abf + (size_t)(m0 + lm) * DD;
  const ushort_t* brow = wo  + (size_t)(n0 + lm) * DD;
  for (int k0 = 0; k0 < DD; k0 += 32) {
    FragAB af, bf_;
    af.q[0]  = *(const u32x4*)(arow + k0 + lg * 8);
    af.q[1]  = *(const u32x4*)(arow + k0 + 16 + lg * 8);
    bf_.q[0] = *(const u32x4*)(brow + k0 + lg * 8);
    bf_.q[1] = *(const u32x4*)(brow + k0 + 16 + lg * 8);
    acc = wmma_bf16(af, bf_, acc);
  }
  const float bias = bo[n0 + lm];
  #pragma unroll
  for (int r = 0; r < 8; ++r) {
    const int row = m0 + r + lg * 8;
    const int bb = row >> 4, t = row & 15;
    const float v = (t < ncnt[bb]) ? (acc[r] + bias) : 0.f;
    out[(size_t)row * DD + n0 + lm] = v;
  }
}

// ---------------------------------------------------------------- launch
extern "C" void kernel_launch(void* const* d_in, const int* in_sizes, int n_in,
                              void* d_out, int out_size, void* d_ws, size_t ws_size,
                              hipStream_t stream)
{
  const float* x    = (const float*)d_in[0];
  const float* rc   = (const float*)d_in[1];
  const float* rs   = (const float*)d_in[2];
  const float* pk   = (const float*)d_in[3];
  const float* pv   = (const float*)d_in[4];
  const float* Wq   = (const float*)d_in[5];
  const float* bq   = (const float*)d_in[6];
  const float* Wk   = (const float*)d_in[7];
  const float* bk   = (const float*)d_in[8];
  const float* Wv   = (const float*)d_in[9];
  const float* bv   = (const float*)d_in[10];
  const float* Wo   = (const float*)d_in[11];
  const float* bo   = (const float*)d_in[12];
  const int*  plen  = (const int*)d_in[13];
  const int*  ncnt  = (const int*)d_in[14];
  (void)in_sizes; (void)n_in; (void)out_size; (void)ws_size;

  char* ws = (char*)d_ws;
  const size_t NX = (size_t)BB * TNN * DD;   // 262144
  const size_t NW = (size_t)DD * DD;         // 1048576
  ushort_t* xbf  = (ushort_t*)ws;
  ushort_t* wqbf = xbf  + NX;
  ushort_t* wkbf = wqbf + NW;
  ushort_t* wvbf = wkbf + NW;
  ushort_t* wobf = wvbf + NW;
  float* qraw = (float*)(wobf + NW);
  float* kraw = qraw + NX;
  float* vraw = kraw + NX;
  ushort_t* attnbf = (ushort_t*)(vraw + NX);

  cvt_kernel<<<(int)((NX / 4 + 255) / 256), 256, 0, stream>>>(x,  xbf,  (int)(NX / 4));
  cvt_kernel<<<(int)((NW / 4 + 255) / 256), 256, 0, stream>>>(Wq, wqbf, (int)(NW / 4));
  cvt_kernel<<<(int)((NW / 4 + 255) / 256), 256, 0, stream>>>(Wk, wkbf, (int)(NW / 4));
  cvt_kernel<<<(int)((NW / 4 + 255) / 256), 256, 0, stream>>>(Wv, wvbf, (int)(NW / 4));
  cvt_kernel<<<(int)((NW / 4 + 255) / 256), 256, 0, stream>>>(Wo, wobf, (int)(NW / 4));

  qkv_gemm_kernel<<<dim3(DD / 16, (BB * TNN) / 16, 3), 32, 0, stream>>>(
      xbf, wqbf, wkbf, wvbf, bq, bk, bv, qraw, kraw, vraw);

  rotary_kernel<<<(BB * HH * TNN * 32 + 255) / 256, 256, 0, stream>>>(qraw, kraw, rc, rs);

  attn_kernel<<<BB * HH, 32, 0, stream>>>(qraw, kraw, vraw, pk, pv, plen, ncnt, attnbf);

  out_gemm_kernel<<<dim3(DD / 16, (BB * TNN) / 16), 32, 0, stream>>>(
      attnbf, wobf, bo, ncnt, (float*)d_out);
}
